// SALAAttention2D_20581483283108
// MI455X (gfx1250) — matrix operand order repaired and checked by reading, not compiled, not run on hardware
//
#include <hip/hip_runtime.h>
#include <hip/hip_bf16.h>

typedef __attribute__((ext_vector_type(16))) __bf16 v16bf;
typedef __attribute__((ext_vector_type(8)))  __bf16 v8bf;
typedef __attribute__((ext_vector_type(8)))  float  v8f;

// ---------------------------------------------------------------------------
// Fragment loaders (wave32, per CDNA5 ISA 7.12.2):
//   A 16x32 bf16 : lane L -> row m=L&15; elems 0..7 = K k0+8*(L>>4)+0..7,
//                  elems 8..15 = K k0+16+8*(L>>4)+0..7   (chunks at +0, +16)
//   B 32x16 bf16 : lane L -> col n=L&15; same K order as A
//                  (chunks at +0, +16 in N-major Bt[N,K])
// ---------------------------------------------------------------------------
__device__ __forceinline__ v16bf load_afrag(const __bf16* p)
{
    const v8bf x0 = *(const v8bf*)(p);
    const v8bf x1 = *(const v8bf*)(p + 16);
    v16bf r;
#pragma unroll
    for (int i = 0; i < 8; ++i) { r[i] = x0[i]; r[i + 8] = x1[i]; }
    return r;
}

__device__ __forceinline__ v16bf load_bfrag(const __bf16* p)
{
    const v8bf x0 = *(const v8bf*)(p);
    const v8bf x1 = *(const v8bf*)(p + 16);
    v16bf r;
#pragma unroll
    for (int i = 0; i < 8; ++i) { r[i] = x0[i]; r[i + 8] = x1[i]; }
    return r;
}

// ---------------------------------------------------------------------------
// Generic batched GEMM:  Out[M,N] = A[M,K] (bf16, row major) * Bt[N,K]^T (bf16,
// N-major "transposed B") + bias.  One wave computes a 32(M) x 64(N) register
// tile: 2 A-frags x 4 B-frags -> 8 V_WMMA_F32_16X16X32_BF16 per 32-wide k-step
// (1.5 b128 loads per WMMA).  Batch (blockIdx.z) decomposes as (b, h).
// Requires M % 32 == 0, N % 64 == 0, K % 32 == 0 (true for every call here).
// ---------------------------------------------------------------------------
__global__ __launch_bounds__(32)
void gemm_bf16_wmma(const __bf16* __restrict__ A, long long aSB, long long aSH, int lda,
                    const __bf16* __restrict__ Bt, long long bSB, long long bSH, int ldb,
                    const float* __restrict__ bias,
                    float* __restrict__ Out, long long oSB, long long oSH, int ldo,
                    int Kdim, int nH)
{
    const int lane = threadIdx.x & 31;
    const int bz = blockIdx.z;
    const int bb = bz / nH, hh = bz % nH;
    A   += bb * aSB + hh * aSH;
    Bt  += bb * bSB + hh * bSH;
    Out += bb * oSB + hh * oSH;

    const int mi  = lane & 15;
    const int kro = lane >> 4;
    const int ka = kro * 8;
    const int kb = kro * 8;

    const int m0 = blockIdx.x * 32;
    const int n0 = blockIdx.y * 64;

    const __bf16* arow[2];
    const __bf16* brow[4];
#pragma unroll
    for (int mt = 0; mt < 2; ++mt) arow[mt] = A + (long long)(m0 + mt * 16 + mi) * lda + ka;
#pragma unroll
    for (int nt = 0; nt < 4; ++nt) brow[nt] = Bt + (long long)(n0 + nt * 16 + mi) * ldb + kb;

    v8f acc[2][4] = {};
    for (int k = 0; k < Kdim; k += 32) {
        v16bf am[2], bn[4];
#pragma unroll
        for (int mt = 0; mt < 2; ++mt) am[mt] = load_afrag(arow[mt] + k);
#pragma unroll
        for (int nt = 0; nt < 4; ++nt) bn[nt] = load_bfrag(brow[nt] + k);
#pragma unroll
        for (int mt = 0; mt < 2; ++mt)
#pragma unroll
            for (int nt = 0; nt < 4; ++nt)
                acc[mt][nt] = __builtin_amdgcn_wmma_f32_16x16x32_bf16(
                    false, am[mt], false, bn[nt], (short)0, acc[mt][nt], false, false);
    }

#pragma unroll
    for (int nt = 0; nt < 4; ++nt) {
        const int outcol = n0 + nt * 16 + mi;
        const float bv = bias ? bias[outcol] : 0.0f;
#pragma unroll
        for (int mt = 0; mt < 2; ++mt)
#pragma unroll
            for (int r = 0; r < 8; ++r) {
                const int mrow = m0 + mt * 16 + kro * 8 + r;
                Out[(long long)mrow * ldo + outcol] = acc[mt][nt][r] + bv;
            }
    }
}

// --------------------------- helper kernels --------------------------------

// dst[N,K] (bf16) = src[K,N]^T (fp32) -- weight pre-transpose so GEMM B loads
// are contiguous b128s.
__global__ __launch_bounds__(256)
void transpose_to_bf16(const float* __restrict__ src, __bf16* __restrict__ dst,
                       int rows /*K*/, int cols /*N*/)
{
    const int i = blockIdx.x * 256 + threadIdx.x;
    if (i >= rows * cols) return;
    const int c = i / rows, r = i - c * rows;
    dst[i] = (__bf16)src[(long long)r * cols + c];
}

// strided fp32 -> bf16 slab copy (for building concat / extracting head slabs)
__global__ __launch_bounds__(256)
void convert_slab(const float* __restrict__ src, int srcLd, int srcColOff,
                  __bf16* __restrict__ dst, int dstLd, int dstColOff,
                  int rows, int cols)
{
    const int i = blockIdx.x * 256 + threadIdx.x;
    if (i >= rows * cols) return;
    const int r = i / cols, c = i - r * cols;
    dst[(long long)r * dstLd + dstColOff + c] =
        (__bf16)src[(long long)r * srcLd + srcColOff + c];
}

// pack up to 4 length-512 bias vectors into one contiguous fp32 array
__global__ __launch_bounds__(256)
void pack_bias(float* __restrict__ dst, const float* p0, const float* p1,
               const float* p2, const float* p3, int total)
{
    const int i = blockIdx.x * 256 + threadIdx.x;
    if (i >= total) return;
    const float* ps[4] = {p0, p1, p2, p3};
    dst[i] = ps[i >> 9][i & 511];
}

// 2x2 avg-pool of z (B,32,32,C) -> anchors (B,256,C) + layernorm over C.
// one block = one anchor token; 256 threads, 2 channels each; LDS tree reduce.
__global__ __launch_bounds__(256)
void pool_layernorm(const float* __restrict__ z, const float* __restrict__ g,
                    const float* __restrict__ be, float* __restrict__ anch,
                    __bf16* __restrict__ anchb)
{
    __shared__ float red[256];
    const int a = blockIdx.x;                // 0..511 = b*256 + na
    const int b = a >> 8, na = a & 255;
    const int ay = na >> 4, ax = na & 15;
    const int t = threadIdx.x;
    const long long zb = (long long)b * 1024 * 512;
    const int p00 = (2 * ay) * 32 + 2 * ax;
    float vals[2];
#pragma unroll
    for (int i = 0; i < 2; ++i) {
        const int c = t + i * 256;
        vals[i] = 0.25f * (z[zb + (long long)p00 * 512 + c]
                         + z[zb + (long long)(p00 + 1) * 512 + c]
                         + z[zb + (long long)(p00 + 32) * 512 + c]
                         + z[zb + (long long)(p00 + 33) * 512 + c]);
    }
    red[t] = vals[0] + vals[1];
    __syncthreads();
    for (int s = 128; s > 0; s >>= 1) { if (t < s) red[t] += red[t + s]; __syncthreads(); }
    const float mean = red[0] * (1.0f / 512.0f);
    __syncthreads();
    const float d0 = vals[0] - mean, d1 = vals[1] - mean;
    red[t] = d0 * d0 + d1 * d1;
    __syncthreads();
    for (int s = 128; s > 0; s >>= 1) { if (t < s) red[t] += red[t + s]; __syncthreads(); }
    const float inv = rsqrtf(red[0] * (1.0f / 512.0f) + 1e-5f);
#pragma unroll
    for (int i = 0; i < 2; ++i) {
        const int c = t + i * 256;
        const float o = (vals[i] - mean) * inv * g[c] + be[c];
        const long long oi = (long long)a * 512 + c;
        anch[oi] = o;
        anchb[oi] = (__bf16)o;
    }
}

// row softmax over 256 anchor logits (scale folded in), fp32 -> bf16 probs
__global__ __launch_bounds__(256)
void softmax_rows(const float* __restrict__ S, __bf16* __restrict__ P, float scale)
{
    __shared__ float red[256];
    const int row = blockIdx.x, t = threadIdx.x;
    const long long base = (long long)row * 256;
    const float v = S[base + t] * scale;
    red[t] = v;
    __syncthreads();
    for (int s = 128; s > 0; s >>= 1) { if (t < s) red[t] = fmaxf(red[t], red[t + s]); __syncthreads(); }
    const float m = red[0];
    __syncthreads();
    const float e = __expf(v - m);
    red[t] = e;
    __syncthreads();
    for (int s = 128; s > 0; s >>= 1) { if (t < s) red[t] += red[t + s]; __syncthreads(); }
    P[base + t] = (__bf16)(e * (1.0f / red[0]));
}

// build vat[b,h][d][na] = bf16(va[b,na,h,d]) for the PV GEMM's N-major B
__global__ __launch_bounds__(256)
void build_vat(const float* __restrict__ Yk, __bf16* __restrict__ vat)
{
    const int i = blockIdx.x * 256 + threadIdx.x;   // 16*64*256
    const int na = i & 255;
    const int rest = i >> 8;
    const int d = rest & 63;
    const int bh = rest >> 6;
    const int b = bh >> 3, h = bh & 7;
    vat[i] = (__bf16)Yk[(long long)(b * 256 + na) * 1024 + 512 + h * 64 + d];
}

// 7x7 zero-padded window attention. One wave per output pixel; lanes split the
// 64-dim head (2 dims/lane); logit via convergent butterfly reduce; online
// softmax (OOB taps contribute logit==0, v==0, matching F.unfold zero pad).
__global__ __launch_bounds__(256)
void window_attn(const float* __restrict__ Y, float* __restrict__ OL,
                 __bf16* __restrict__ cat)
{
    const int lane = threadIdx.x & 31;
    const int w = threadIdx.x >> 5;
    const int p = blockIdx.x * 8 + w;        // 0..16383 = ((b*8+h)*1024 + n)
    const int bh = p >> 10, n = p & 1023;
    const int b = bh >> 3, h = bh & 7;
    const int y = n >> 5, x = n & 31;
    const int d0 = lane * 2;
    const long long qrow = ((long long)(b * 1024 + n)) * 2048 + h * 64;
    const float q0 = Y[qrow + d0], q1 = Y[qrow + d0 + 1];
    float mv = -1e30f, ss = 0.f, o0 = 0.f, o1 = 0.f;
    for (int j = 0; j < 49; ++j) {
        const int dy = j / 7 - 3, dx = j % 7 - 3;
        const int yy = y + dy, xx = x + dx;
        float l = 0.f, v0 = 0.f, v1 = 0.f;
        if (yy >= 0 && yy < 32 && xx >= 0 && xx < 32) {   // uniform per wave
            const long long kr = ((long long)(b * 1024 + yy * 32 + xx)) * 2048 + h * 64;
            const float k0 = Y[kr + 512 + d0], k1 = Y[kr + 512 + d0 + 1];
            v0 = Y[kr + 1024 + d0]; v1 = Y[kr + 1024 + d0 + 1];
            float part = q0 * k0 + q1 * k1;
#pragma unroll
            for (int msk = 16; msk >= 1; msk >>= 1) part += __shfl_xor(part, msk, 32);
            l = part * 0.125f;                 // SCALE = 64^-0.5
        }
        const float mn = fmaxf(mv, l);
        const float corr = __expf(mv - mn);
        const float e = __expf(l - mn);
        ss = ss * corr + e;
        o0 = o0 * corr + e * v0;
        o1 = o1 * corr + e * v1;
        mv = mn;
    }
    const float inv = 1.0f / ss;
    const float r0 = o0 * inv, r1 = o1 * inv;
    const long long orow = (long long)(b * 1024 + n);
    OL[orow * 512 + h * 64 + d0]     = r0;
    OL[orow * 512 + h * 64 + d0 + 1] = r1;
    cat[orow * 1536 + 512 + h * 64 + d0]     = (__bf16)r0;
    cat[orow * 1536 + 512 + h * 64 + d0 + 1] = (__bf16)r1;
}

// F = o_loc + sigmoid(G) * o_anc, written bf16 for the final WMMA GEMM
__global__ __launch_bounds__(256)
void fuse_gate(const float* __restrict__ OL, const float* __restrict__ OA,
               const float* __restrict__ G, __bf16* __restrict__ Fb)
{
    const int i = blockIdx.x * 256 + threadIdx.x;   // 2048*512 exact
    const float g = 1.0f / (1.0f + __expf(-G[i]));
    Fb[i] = (__bf16)(OL[i] + g * OA[i]);
}

// ---------------------------------------------------------------------------
extern "C" void kernel_launch(void* const* d_in, const int* in_sizes, int n_in,
                              void* d_out, int out_size, void* d_ws, size_t ws_size,
                              hipStream_t stream)
{
    (void)in_sizes; (void)n_in; (void)out_size; (void)ws_size;
    const float* x      = (const float*)d_in[0];
    const float* z      = (const float*)d_in[1];
    const float* wq_loc = (const float*)d_in[2];   const float* bq_loc = (const float*)d_in[3];
    const float* wk_loc = (const float*)d_in[4];   const float* bk_loc = (const float*)d_in[5];
    const float* wv_loc = (const float*)d_in[6];   const float* bv_loc = (const float*)d_in[7];
    const float* wq_anc = (const float*)d_in[8];   const float* bq_anc = (const float*)d_in[9];
    const float* wk_anc = (const float*)d_in[10];  const float* bk_anc = (const float*)d_in[11];
    const float* wv_anc = (const float*)d_in[12];  const float* bv_anc = (const float*)d_in[13];
    const float* wg     = (const float*)d_in[14];  const float* bg     = (const float*)d_in[15];
    const float* wo     = (const float*)d_in[16];  const float* bo     = (const float*)d_in[17];
    const float* ln_g   = (const float*)d_in[18];  const float* ln_b   = (const float*)d_in[19];
    float* out = (float*)d_out;

    char* ws = (char*)d_ws;
    size_t off = 0;
    auto carve = [&](size_t bytes) -> void* {
        void* pp = ws + off;
        off = (off + bytes + 255) & ~(size_t)255;
        return pp;
    };
    __bf16* cat   = (__bf16*)carve(2048ull * 1536 * 2);  // [x | o_loc | o_anc] bf16
    __bf16* Wt1   = (__bf16*)carve(2048ull * 512 * 2);   // [wq,wk,wv,wq_anc]^T
    __bf16* Wt2   = (__bf16*)carve(1024ull * 512 * 2);   // [wk_anc,wv_anc]^T
    __bf16* Wtg   = (__bf16*)carve(512ull * 1536 * 2);   // wg^T
    __bf16* Wto   = (__bf16*)carve(512ull * 512 * 2);    // wo^T
    float*  bias1 = (float*)carve(2048 * 4);
    float*  bias2 = (float*)carve(1024 * 4);
    float*  Y     = (float*)carve(2048ull * 2048 * 4);   // fused q|k|v|qa
    float*  anch  = (float*)carve(512ull * 512 * 4);
    __bf16* anchb = (__bf16*)carve(512ull * 512 * 2);
    float*  Yk    = (float*)carve(512ull * 1024 * 4);    // fused ka|va
    __bf16* qab   = (__bf16*)carve(2048ull * 512 * 2);
    __bf16* kab   = (__bf16*)carve(512ull * 512 * 2);
    __bf16* vat   = (__bf16*)carve(16ull * 64 * 256 * 2);
    float*  S     = (float*)carve(16ull * 1024 * 256 * 4);
    __bf16* P     = (__bf16*)carve(16ull * 1024 * 256 * 2);
    float*  OL    = (float*)carve(2048ull * 512 * 4);
    float*  OA    = (float*)carve(2048ull * 512 * 4);
    float*  G     = (float*)carve(2048ull * 512 * 4);
    __bf16* Fb    = (__bf16*)carve(2048ull * 512 * 2);

    // 1) weight pre-transpose to bf16 [N,K] + bias packs
    transpose_to_bf16<<<1024, 256, 0, stream>>>(wq_loc, Wt1,              512, 512);
    transpose_to_bf16<<<1024, 256, 0, stream>>>(wk_loc, Wt1 + 512 * 512,  512, 512);
    transpose_to_bf16<<<1024, 256, 0, stream>>>(wv_loc, Wt1 + 1024 * 512, 512, 512);
    transpose_to_bf16<<<1024, 256, 0, stream>>>(wq_anc, Wt1 + 1536 * 512, 512, 512);
    transpose_to_bf16<<<1024, 256, 0, stream>>>(wk_anc, Wt2,              512, 512);
    transpose_to_bf16<<<1024, 256, 0, stream>>>(wv_anc, Wt2 + 512 * 512,  512, 512);
    transpose_to_bf16<<<3072, 256, 0, stream>>>(wg,     Wtg,             1536, 512);
    transpose_to_bf16<<<1024, 256, 0, stream>>>(wo,     Wto,              512, 512);
    pack_bias<<<8, 256, 0, stream>>>(bias1, bq_loc, bk_loc, bv_loc, bq_anc, 2048);
    pack_bias<<<4, 256, 0, stream>>>(bias2, bk_anc, bv_anc, bk_anc, bk_anc, 1024);

    // 2) x -> bf16 (doubles as concat slab 0)
    convert_slab<<<4096, 256, 0, stream>>>(x, 512, 0, cat, 1536, 0, 2048, 512);

    // 3) fused projection GEMM: Y[2048,2048] = x @ [wq|wk|wv|wq_anc] + bias
    gemm_bf16_wmma<<<dim3(64, 32, 1), 32, 0, stream>>>(
        cat, 0, 0, 1536, Wt1, 0, 0, 512, bias1, Y, 0, 0, 2048, 512, 1);

    // 4) anchors: pool + layernorm
    pool_layernorm<<<512, 256, 0, stream>>>(z, ln_g, ln_b, anch, anchb);

    // 5) Yk[512,1024] = anchors @ [wk_anc|wv_anc] + bias
    gemm_bf16_wmma<<<dim3(16, 16, 1), 32, 0, stream>>>(
        anchb, 0, 0, 512, Wt2, 0, 0, 512, bias2, Yk, 0, 0, 1024, 512, 1);

    // 6) bf16 operand slabs for attention GEMMs
    convert_slab<<<4096, 256, 0, stream>>>(Y, 2048, 1536, qab, 512, 0, 2048, 512);
    convert_slab<<<1024, 256, 0, stream>>>(Yk, 1024, 0, kab, 512, 0, 512, 512);
    build_vat<<<1024, 256, 0, stream>>>(Yk, vat);

    // 7) anchor scores S[bh,1024,256] = qa @ ka^T  (batched per b,h)
    gemm_bf16_wmma<<<dim3(32, 4, 16), 32, 0, stream>>>(
        qab, 1024ll * 512, 64, 512,
        kab, 256ll * 512, 64, 512, nullptr,
        S, 8ll * 1024 * 256, 1024ll * 256, 256, 64, 8);

    // 8) softmax (scale 0.125) -> bf16 probs
    softmax_rows<<<16384, 256, 0, stream>>>(S, P, 0.125f);

    // 9) o_anc[2048,512] = P @ va  (batched per b,h, scattered into head cols)
    gemm_bf16_wmma<<<dim3(32, 1, 16), 32, 0, stream>>>(
        P, 8ll * 1024 * 256, 1024ll * 256, 256,
        vat, 8ll * 64 * 256, 64ll * 256, 256, nullptr,
        OA, 1024ll * 512, 64, 512, 256, 8);

    // 10) local 7x7 window attention -> OL fp32 + concat slab 1 (bf16)
    window_attn<<<2048, 256, 0, stream>>>(Y, OL, cat);

    // 11) o_anc -> concat slab 2 (bf16)
    convert_slab<<<4096, 256, 0, stream>>>(OA, 512, 0, cat, 1536, 1024, 2048, 512);

    // 12) gate logits G = concat @ wg + bg
    gemm_bf16_wmma<<<dim3(64, 8, 1), 32, 0, stream>>>(
        cat, 0, 0, 1536, Wtg, 0, 0, 1536, bg, G, 0, 0, 512, 1536, 1);

    // 13) F = o_loc + sigmoid(G) * o_anc  (bf16)
    fuse_gate<<<4096, 256, 0, stream>>>(OL, OA, G, Fb);

    // 14) out = F @ wo + bo
    gemm_bf16_wmma<<<dim3(64, 8, 1), 32, 0, stream>>>(
        Fb, 0, 0, 512, Wto, 0, 0, 512, bo, out, 0, 0, 512, 512, 1);
}
